// graph_convolution_33938831573463
// MI455X (gfx1250) — matrix operand order repaired and checked by reading, not compiled, not run on hardware
//
#include <hip/hip_runtime.h>

#define N_NODES 50000
#define N_EDGES 800000
#define NL 3
#define K1 160      // msg FFN1 K padded 144 -> 160 (5 x 32)
#define LDA 168     // LDS row stride (halfs) for 160-wide rows (16B aligned)
#define LDM 72      // LDS row stride (halfs) for 64-wide m1 rows
#define LDU 136     // LDS row stride (halfs) for 128-wide update rows
#define EPS 1e-5f

typedef __attribute__((ext_vector_type(16))) _Float16 v16h;
typedef __attribute__((ext_vector_type(8)))  _Float16 v8h;
typedef __attribute__((ext_vector_type(2)))  _Float16 v2h;
typedef __attribute__((ext_vector_type(8)))  float    v8f;

union AF { v16h v; v8h p[2]; };

// Bias add + LayerNorm(64) + ReLU on a 16x64 tile held in 4 WMMA f32 accumulators.
// Accumulator layout: acc[t][v] = element (row = v + 8*(lane>=16), col = t*16 + (lane&15)).
// Row reduction = in-lane sum over t, then shfl_xor tree over the 16-lane half.
__device__ __forceinline__ void ln_relu(v8f* c, const float* __restrict__ bias, int mcol) {
#pragma unroll
  for (int t = 0; t < 4; ++t) {
    float bb = bias[t * 16 + mcol];
#pragma unroll
    for (int v = 0; v < 8; ++v) c[t][v] += bb;
  }
#pragma unroll
  for (int v = 0; v < 8; ++v) {
    float s1 = 0.f, s2 = 0.f;
#pragma unroll
    for (int t = 0; t < 4; ++t) { float y = c[t][v]; s1 += y; s2 += y * y; }
#pragma unroll
    for (int off = 1; off < 16; off <<= 1) {   // stays within 16-lane half (wave32)
      s1 += __shfl_xor(s1, off, 32);
      s2 += __shfl_xor(s2, off, 32);
    }
    float mean = s1 * (1.0f / 64.0f);
    float var  = s2 * (1.0f / 64.0f) - mean * mean;
    float rs = rsqrtf(var + EPS);
#pragma unroll
    for (int t = 0; t < 4; ++t) {
      float y = (c[t][v] - mean) * rs;
      c[t][v] = y > 0.f ? y : 0.f;
    }
  }
}

// Pre-transpose weights to f16 W^T [N][K] so each lane's B-fragment is a
// contiguous 32B load (B order chosen here, same as A: lane = col n, K in contiguous 16-chunks per half).
__global__ void prep_weights(const float* __restrict__ w1, const float* __restrict__ w2,
                             const float* __restrict__ wu,
                             _Float16* __restrict__ w1t, _Float16* __restrict__ w2t,
                             _Float16* __restrict__ wut) {
  int i = blockIdx.x * 256 + threadIdx.x;
  if (i < NL * 64 * K1) {        // w1t[l][n][k], k<144 real else 0
    int l = i / (64 * K1), r = i % (64 * K1), n = r / K1, k = r % K1;
    w1t[i] = (_Float16)((k < 144) ? w1[((size_t)l * 144 + k) * 64 + n] : 0.f);
  }
  if (i < NL * 64 * 64) {
    int l = i / (64 * 64), r = i % (64 * 64), n = r / 64, k = r % 64;
    w2t[i] = (_Float16)w2[((size_t)l * 64 + k) * 64 + n];
  }
  if (i < NL * 64 * 128) {
    int l = i / (64 * 128), r = i % (64 * 128), n = r / 128, k = r % 128;
    wut[i] = (_Float16)wu[((size_t)l * 128 + k) * 64 + n];
  }
}

__global__ void zero_kernel(float4* __restrict__ p, int n) {
  int i = blockIdx.x * 256 + threadIdx.x;
  if (i < n) p[i] = make_float4(0.f, 0.f, 0.f, 0.f);
}

// One wave per 16-edge tile: gather -> GEMM1(16x160x64) -> LN/ReLU -> GEMM2(16x64x64)
// -> LN/ReLU -> atomic scatter-add into agg (L2-resident).
__global__ __launch_bounds__(128) void edge_msg_kernel(
    const float* __restrict__ x, const float* __restrict__ ef,
    const int* __restrict__ srcI, const int* __restrict__ dstI,
    const _Float16* __restrict__ w1t, const float* __restrict__ b1,
    const _Float16* __restrict__ w2t, const float* __restrict__ b2,
    float* __restrict__ agg) {
  __shared__ __align__(16) _Float16 sA[4][16 * LDA];
  __shared__ __align__(16) _Float16 sM[4][16 * LDM];
  __shared__ int sD[4][16];

  const int wv = threadIdx.x >> 5;
  const int lane = threadIdx.x & 31;
  const int mcol = lane & 15;
  const int hi = lane >> 4;
  const int tile = blockIdx.x * 4 + wv;   // grid exact: 50000 tiles / 4 waves
  const int e0 = tile * 16;

  _Float16* A = sA[wv];
  _Float16* M = sM[wv];
  int* D = sD[wv];

  if (lane < 16) D[lane] = dstI[e0 + lane];

  // gather: row = [x[dst](64) | x[src](64) | edge(16) | zeros(16)] as f16
#pragma unroll 1
  for (int r = 0; r < 16; ++r) {
    int e = e0 + r;
    int dN = dstI[e];
    int sN = srcI[e];
    _Float16* row = A + r * LDA;
    float2 vd = ((const float2*)(x + (size_t)dN * 64))[lane];
    v2h h0; h0.x = (_Float16)vd.x; h0.y = (_Float16)vd.y;
    *(v2h*)(row + 2 * lane) = h0;
    float2 vs = ((const float2*)(x + (size_t)sN * 64))[lane];
    v2h h1; h1.x = (_Float16)vs.x; h1.y = (_Float16)vs.y;
    *(v2h*)(row + 64 + 2 * lane) = h1;
    if (lane < 8) {
      float2 ve = ((const float2*)(ef + (size_t)e * 16))[lane];
      v2h h2; h2.x = (_Float16)ve.x; h2.y = (_Float16)ve.y;
      *(v2h*)(row + 128 + 2 * lane) = h2;
      v2h hz; hz.x = (_Float16)0.f; hz.y = (_Float16)0.f;
      *(v2h*)(row + 144 + 2 * lane) = hz;
    }
  }
  __syncthreads();

  // GEMM1: [16x160] x [160x64]
  v8f c[4];
#pragma unroll
  for (int t = 0; t < 4; ++t) { v8f z = {}; c[t] = z; }
#pragma unroll
  for (int kt = 0; kt < 5; ++kt) {
    AF a;
    const _Float16* ar = A + mcol * LDA + kt * 32 + hi * 16;
    a.p[0] = *(const v8h*)(ar);
    a.p[1] = *(const v8h*)(ar + 8);
#pragma unroll
    for (int t = 0; t < 4; ++t) {
      const v16h b = *(const v16h*)(w1t + (size_t)(t * 16 + mcol) * K1 + kt * 32 + hi * 16);
      c[t] = __builtin_amdgcn_wmma_f32_16x16x32_f16(false, a.v, false, b, (short)0, c[t], false, false);
    }
  }
  ln_relu(c, b1, mcol);

  // m1 accumulators -> LDS (f16, row-major) to rebuild A-fragments
#pragma unroll
  for (int t = 0; t < 4; ++t)
#pragma unroll
    for (int v = 0; v < 8; ++v)
      M[(v + 8 * hi) * LDM + t * 16 + mcol] = (_Float16)c[t][v];
  __syncthreads();

  // GEMM2: [16x64] x [64x64]
  v8f d2[4];
#pragma unroll
  for (int t = 0; t < 4; ++t) { v8f z = {}; d2[t] = z; }
#pragma unroll
  for (int kt = 0; kt < 2; ++kt) {
    AF a;
    const _Float16* ar = M + mcol * LDM + kt * 32 + hi * 16;
    a.p[0] = *(const v8h*)(ar);
    a.p[1] = *(const v8h*)(ar + 8);
#pragma unroll
    for (int t = 0; t < 4; ++t) {
      const v16h b = *(const v16h*)(w2t + (size_t)(t * 16 + mcol) * 64 + kt * 32 + hi * 16);
      d2[t] = __builtin_amdgcn_wmma_f32_16x16x32_f16(false, a.v, false, b, (short)0, d2[t], false, false);
    }
  }
  ln_relu(d2, b2, mcol);

  // scatter-add into agg[dst] (f32 atomics, L2-resident target)
#pragma unroll
  for (int v = 0; v < 8; ++v) {
    int node = D[v + 8 * hi];
    float* ap = agg + (size_t)node * 64 + mcol;
#pragma unroll
    for (int t = 0; t < 4; ++t) atomicAdd(ap + t * 16, d2[t][v]);
  }
}

// One wave per 16-node tile: [x|agg] (16x128) x WuT -> LN/ReLU -> residual.
// Safe in-place (x == xout): each tile reads its rows before writing them.
__global__ __launch_bounds__(32) void node_upd_kernel(
    const float* __restrict__ x, const float* __restrict__ agg,
    const _Float16* __restrict__ wut, const float* __restrict__ bu,
    float* __restrict__ xout) {
  __shared__ __align__(16) _Float16 A[16 * LDU];
  const int lane = threadIdx.x;
  const int mcol = lane & 15;
  const int hi = lane >> 4;
  const int n0 = blockIdx.x * 16;

#pragma unroll 1
  for (int r = 0; r < 16; ++r) {
    int n = n0 + r;
    _Float16* row = A + r * LDU;
    float2 vx = ((const float2*)(x + (size_t)n * 64))[lane];
    v2h h0; h0.x = (_Float16)vx.x; h0.y = (_Float16)vx.y;
    *(v2h*)(row + 2 * lane) = h0;
    float2 va = ((const float2*)(agg + (size_t)n * 64))[lane];
    v2h h1; h1.x = (_Float16)va.x; h1.y = (_Float16)va.y;
    *(v2h*)(row + 64 + 2 * lane) = h1;
  }
  __syncthreads();

  v8f c[4];
#pragma unroll
  for (int t = 0; t < 4; ++t) { v8f z = {}; c[t] = z; }
#pragma unroll
  for (int kt = 0; kt < 4; ++kt) {
    AF a;
    const _Float16* ar = A + mcol * LDU + kt * 32 + hi * 16;
    a.p[0] = *(const v8h*)(ar);
    a.p[1] = *(const v8h*)(ar + 8);
#pragma unroll
    for (int t = 0; t < 4; ++t) {
      const v16h b = *(const v16h*)(wut + (size_t)(t * 16 + mcol) * 128 + kt * 32 + hi * 16);
      c[t] = __builtin_amdgcn_wmma_f32_16x16x32_f16(false, a.v, false, b, (short)0, c[t], false, false);
    }
  }
  ln_relu(c, bu, mcol);

#pragma unroll
  for (int v = 0; v < 8; ++v) {
    int n = n0 + v + 8 * hi;
    const float* xr = x + (size_t)n * 64 + mcol;
    float* orow = xout + (size_t)n * 64 + mcol;
#pragma unroll
    for (int t = 0; t < 4; ++t) orow[t * 16] = xr[t * 16] + c[t][v];
  }
}

extern "C" void kernel_launch(void* const* d_in, const int* in_sizes, int n_in,
                              void* d_out, int out_size, void* d_ws, size_t ws_size,
                              hipStream_t stream) {
  (void)in_sizes; (void)n_in; (void)out_size; (void)ws_size;
  const float* xin = (const float*)d_in[0];
  const float* ef  = (const float*)d_in[1];
  const int*   ei  = (const int*)d_in[2];
  const float* w1  = (const float*)d_in[3];
  const float* b1  = (const float*)d_in[4];
  const float* w2  = (const float*)d_in[5];
  const float* b2  = (const float*)d_in[6];
  const float* wu  = (const float*)d_in[7];
  const float* bu  = (const float*)d_in[8];
  float* out = (float*)d_out;

  const int* srcI = ei;             // edge_index[0] = source (x_j)
  const int* dstI = ei + N_EDGES;   // edge_index[1] = target (x_i / aggregation)

  // workspace: xA (12.8MB) | agg (12.8MB) | f16 weights (~135KB)
  char* ws = (char*)d_ws;
  float* xA  = (float*)ws;
  float* agg = (float*)(ws + (size_t)N_NODES * 64 * 4);
  _Float16* w1t = (_Float16*)(ws + (size_t)N_NODES * 64 * 4 * 2);
  _Float16* w2t = w1t + (size_t)NL * 64 * K1;
  _Float16* wut = w2t + (size_t)NL * 64 * 64;

  prep_weights<<<120, 256, 0, stream>>>(w1, w2, wu, w1t, w2t, wut);

  const float* cur = xin;
  for (int l = 0; l < NL; ++l) {
    zero_kernel<<<(N_NODES * 64 / 4 + 255) / 256, 256, 0, stream>>>((float4*)agg, N_NODES * 64 / 4);
    edge_msg_kernel<<<N_EDGES / 16 / 4, 128, 0, stream>>>(
        cur, ef, srcI, dstI,
        w1t + (size_t)l * 64 * K1, b1 + l * 64,
        w2t + (size_t)l * 64 * 64, b2 + l * 64, agg);
    float* nxt = (l == NL - 1) ? out : xA;
    node_upd_kernel<<<N_NODES / 16, 32, 0, stream>>>(
        cur, agg, wut + (size_t)l * 64 * 128, bu + l * 64, nxt);
    cur = nxt;
  }
}
